// SetAbstraction_89438398972531
// MI455X (gfx1250) — compile-verified, matrix operand order checked by reading, not run on hardware
//
#include <hip/hip_runtime.h>
#include <hip/hip_bf16.h>

typedef __attribute__((ext_vector_type(16))) _Float16 v16h;
typedef __attribute__((ext_vector_type(8)))  _Float16 v8h;
typedef __attribute__((ext_vector_type(4)))  _Float16 v4h;
typedef __attribute__((ext_vector_type(8)))  float    v8f;

#define BQ   8
#define PQ   2048
#define NQ   (BQ * PQ)          // 16384
#define DIN  128
#define KN   32                 // max neighbors
#define HD   256                // hidden
#define DOUTQ 256
#define K1   131                // DIN + 3
#define K1P  160                // padded to multiple of 32
#define MS   168                // msg LDS row stride (f16), padded for banks
#define HS   264                // h1  LDS row stride (f16), padded for banks
#define PTS  8                  // points per workgroup
#define ROWSQ (PTS * KN)        // 256 message rows
#define R2Q  0.04f

union HV { v16h v; v8h h[2]; };

static __device__ __forceinline__ v16h ld16(const _Float16* p0, const _Float16* p1) {
    HV u;
    u.h[0] = *(const v8h*)p0;
    u.h[1] = *(const v8h*)p1;
    return u.v;
}

// ---------------------------------------------------------------------------
// Kernel 1: convert weights to f16, transpose to column-major, zero-pad K1->K1P
// W1p[c*K1P + k] = (k<K1) ? f16(W1[k*HD + c]) : 0      (c in [0,HD))
// W2p[c*HD  + k] = f16(W2[k*DOUTQ + c])                (c in [0,DOUTQ))
// ---------------------------------------------------------------------------
__global__ void prep_weights(const float* __restrict__ W1, const float* __restrict__ W2,
                             _Float16* __restrict__ W1p, _Float16* __restrict__ W2p) {
    int idx = blockIdx.x * blockDim.x + threadIdx.x;
    const int n1 = HD * K1P;
    if (idx < n1) {
        int c = idx / K1P, k = idx % K1P;
        float v = (k < K1) ? W1[k * HD + c] : 0.0f;
        W1p[idx] = (_Float16)v;
    } else if (idx < n1 + DOUTQ * HD) {
        int t = idx - n1;
        int c = t / HD, k = t % HD;
        W2p[t] = (_Float16)W2[k * DOUTQ + c];
    }
}

// ---------------------------------------------------------------------------
// Kernel 2: ball query, first KN neighbors in ascending index order per point
// ---------------------------------------------------------------------------
__global__ void ball_query(const float* __restrict__ pos,
                           int* __restrict__ nbr, int* __restrict__ cnt) {
    __shared__ float sp[256 * 3];
    const int tid = threadIdx.x;
    const int bpc = PQ / 256;                       // blocks per cloud = 8
    const int b  = blockIdx.x / bpc;
    const int ib = (blockIdx.x % bpc) * 256 + tid;  // point-in-cloud
    const int i  = b * PQ + ib;
    const float px = pos[3 * i], py = pos[3 * i + 1], pz = pos[3 * i + 2];
    int count = 0;
    int* my = nbr + (size_t)i * KN;
    for (int jt = 0; jt < PQ; jt += 256) {
        __syncthreads();
        const int js = b * PQ + jt + tid;
        sp[3 * tid]     = pos[3 * js];
        sp[3 * tid + 1] = pos[3 * js + 1];
        sp[3 * tid + 2] = pos[3 * js + 2];
        __syncthreads();
        for (int jj = 0; jj < 256; ++jj) {
            float dx = sp[3 * jj] - px;
            float dy = sp[3 * jj + 1] - py;
            float dz = sp[3 * jj + 2] - pz;
            float d2 = dx * dx + dy * dy + dz * dz;
            if (d2 <= R2Q && count < KN) { my[count] = jt + jj; ++count; }
        }
    }
    cnt[i] = count;
}

// ---------------------------------------------------------------------------
// Kernel 3: fused gather + MLP (WMMA f16) + max aggregation.
// One workgroup (256 thr = 8 waves) handles PTS=8 points (256 rows).
// ---------------------------------------------------------------------------
__global__ __launch_bounds__(256, 1) void sa_fused(
    const float* __restrict__ x, const float* __restrict__ pos,
    const int* __restrict__ nbr, const int* __restrict__ cnt,
    const _Float16* __restrict__ W1p, const float* __restrict__ b1,
    const _Float16* __restrict__ W2p, const float* __restrict__ b2,
    float* __restrict__ out)
{
    extern __shared__ char smem[];
    _Float16* msg = (_Float16*)smem;                              // ROWSQ x MS
    _Float16* h1  = (_Float16*)(smem + ROWSQ * MS * 2);           // ROWSQ x HS
    int* scnt     = (int*)(smem + ROWSQ * MS * 2 + ROWSQ * HS * 2);

    const int tid  = threadIdx.x;
    const int lane = tid & 31;
    const int wave = tid >> 5;
    const int p0   = blockIdx.x * PTS;

    if (tid < PTS) scnt[tid] = cnt[p0 + tid];
    __syncthreads();

    // ---- Phase A: gather messages into LDS as f16 --------------------------
    for (int row = wave; row < ROWSQ; row += 8) {
        const int pt = row >> 5, k = row & 31;
        const int gpt = p0 + pt;
        const bool valid = k < scnt[pt];
        const int cloudBase = (gpt / PQ) * PQ;
        const int j   = valid ? nbr[(size_t)gpt * KN + k] : 0;
        const int src = cloudBase + j;
        float4 xv = make_float4(0.f, 0.f, 0.f, 0.f);
        if (valid) xv = ((const float4*)x)[src * (DIN / 4) + lane];
        v4h hv;
        hv[0] = (_Float16)xv.x; hv[1] = (_Float16)xv.y;
        hv[2] = (_Float16)xv.z; hv[3] = (_Float16)xv.w;
        *(v4h*)(msg + row * MS + lane * 4) = hv;
        float rv = 0.f;
        if (valid && lane < 3) rv = pos[src * 3 + lane] - pos[gpt * 3 + lane];
        msg[row * MS + DIN + lane] = (_Float16)rv;   // rel pos + zero pad to K1P
    }
    __syncthreads();

    const int half = lane >> 4;          // lane half selects K sub-chunks
    const int l15  = lane & 15;
    const int kOff0 = half ? 8 : 0;      // 16-bit A/B layout: lanes<16 -> k 0-7,16-23
    const int kOff1 = kOff0 + 16;        //                    lanes>=16 -> k 8-15,24-31

    // ---- Phase B: layer 1, msg[256 x 160] @ W1p -> h1[256 x 256] (f16 LDS) -
    for (int cc = 0; cc < 2; ++cc) {
        const int colBase = wave * 32 + cc * 16;
        const int col = colBase + l15;
        v8f acc[16] = {};
        for (int kt = 0; kt < K1P / 32; ++kt) {
            const int kb = kt * 32;
            const _Float16* wp = W1p + col * K1P + kb;
            const v16h bv = ld16(wp + kOff0, wp + kOff1);
#pragma unroll
            for (int rt = 0; rt < 16; ++rt) {
                const _Float16* ap = msg + (rt * 16 + l15) * MS + kb;
                const v16h av = ld16(ap + kOff0, ap + kOff1);
                acc[rt] = __builtin_amdgcn_wmma_f32_16x16x32_f16(
                    false, av, false, bv, (short)0, acc[rt], false, false);
            }
        }
        const float bias = b1[col];
#pragma unroll
        for (int rt = 0; rt < 16; ++rt) {
#pragma unroll
            for (int r = 0; r < 8; ++r) {
                float v = acc[rt][r] + bias;
                v = v > 0.f ? v : 0.f;
                h1[(rt * 16 + r + 8 * half) * HS + col] = (_Float16)v;
            }
        }
    }
    __syncthreads();

    // ---- Phase C: layer 2 + bias + relu + masked max over neighbors --------
    for (int cc = 0; cc < 2; ++cc) {
        const int colBase = wave * 32 + cc * 16;
        const int col = colBase + l15;
        v8f acc[16] = {};
        for (int kt = 0; kt < HD / 32; ++kt) {
            const int kb = kt * 32;
            const _Float16* wp = W2p + col * HD + kb;
            const v16h bv = ld16(wp + kOff0, wp + kOff1);
#pragma unroll
            for (int rt = 0; rt < 16; ++rt) {
                const _Float16* ap = h1 + (rt * 16 + l15) * HS + kb;
                const v16h av = ld16(ap + kOff0, ap + kOff1);
                acc[rt] = __builtin_amdgcn_wmma_f32_16x16x32_f16(
                    false, av, false, bv, (short)0, acc[rt], false, false);
            }
        }
        const float bias = b2[col];
#pragma unroll
        for (int pt = 0; pt < PTS; ++pt) {
            const int c = scnt[pt];
            float m = -3.4e38f;
#pragma unroll
            for (int hh = 0; hh < 2; ++hh) {
                const int rt = pt * 2 + hh;
#pragma unroll
                for (int r = 0; r < 8; ++r) {
                    const int k = hh * 16 + r + 8 * half;   // neighbor slot
                    float v = acc[rt][r] + bias;
                    v = v > 0.f ? v : 0.f;
                    if (k < c) m = fmaxf(m, v);
                }
            }
            const float o = __shfl_xor(m, 16, 32);          // merge lane halves
            m = fmaxf(m, o);
            if (lane < 16) out[(size_t)(p0 + pt) * DOUTQ + colBase + l15] = m;
        }
    }
}

// ---------------------------------------------------------------------------
// Kernel 4: tuple tail — copy pos (f32) and batch (i64) behind `out`
// ---------------------------------------------------------------------------
__global__ void tail_copy(const float* __restrict__ pos,
                          const long long* __restrict__ batch,
                          float* __restrict__ out) {
    const int idx = blockIdx.x * blockDim.x + threadIdx.x;
    if (idx < NQ * 3) {
        out[(size_t)NQ * DOUTQ + idx] = pos[idx];
    } else if (idx < NQ * 3 + NQ) {
        long long* ob = (long long*)(out + (size_t)NQ * DOUTQ + (size_t)NQ * 3);
        ob[idx - NQ * 3] = batch[idx - NQ * 3];
    }
}

extern "C" void kernel_launch(void* const* d_in, const int* in_sizes, int n_in,
                              void* d_out, int out_size, void* d_ws, size_t ws_size,
                              hipStream_t stream) {
    const float*     x     = (const float*)d_in[0];
    const float*     pos   = (const float*)d_in[1];
    const long long* batch = (const long long*)d_in[2];
    const float*     W1    = (const float*)d_in[3];
    const float*     b1    = (const float*)d_in[4];
    const float*     W2    = (const float*)d_in[5];
    const float*     b2    = (const float*)d_in[6];
    float* out = (float*)d_out;

    char* ws = (char*)d_ws;
    _Float16* W1p = (_Float16*)ws;                                // 81920 B
    _Float16* W2p = (_Float16*)(ws + 81920);                      // 131072 B
    int* nbr = (int*)(ws + 81920 + 131072);                       // 2 MB
    int* cnt = (int*)(ws + 81920 + 131072 + (size_t)NQ * KN * 4); // 64 KB

    const int nprep = HD * K1P + DOUTQ * HD;
    prep_weights<<<(nprep + 255) / 256, 256, 0, stream>>>(W1, W2, W1p, W2p);
    ball_query<<<NQ / 256, 256, 0, stream>>>(pos, nbr, cnt);

    const size_t smem = (size_t)ROWSQ * MS * 2 + (size_t)ROWSQ * HS * 2 + PTS * 4;
    hipFuncSetAttribute((const void*)sa_fused,
                        hipFuncAttributeMaxDynamicSharedMemorySize, (int)smem);
    sa_fused<<<NQ / PTS, 256, smem, stream>>>(x, pos, nbr, cnt, W1p, b1, W2p, b2, out);

    tail_copy<<<(NQ * 4 + 255) / 256, 256, 0, stream>>>(pos, batch, out);
}
